// MultiHeadAttention_14637248545292
// MI455X (gfx1250) — matrix operand order repaired and checked by reading, not compiled, not run on hardware
//
#include <hip/hip_runtime.h>

#define HEADS 8
#define DM 256
#define DK 32
#define BB 2
#define SS 2048
#define QSCALE 0.17677669529663687f   // 1/sqrt(32)

typedef __attribute__((ext_vector_type(16))) __bf16        v16bf;
typedef __attribute__((ext_vector_type(8)))  float         v8f;
typedef __attribute__((ext_vector_type(8)))  unsigned int  v8u;

// fp32 -> bf16 round-to-nearest-even
__device__ __forceinline__ unsigned short f2bf(float f) {
  unsigned int u = __builtin_bit_cast(unsigned int, f);
  u += 0x7fffu + ((u >> 16) & 1u);
  return (unsigned short)(u >> 16);
}

__device__ __forceinline__ v16bf mkfrag(const unsigned int* u) {
  v8u t;
#pragma unroll
  for (int i = 0; i < 8; ++i) t[i] = u[i];
  return __builtin_bit_cast(v16bf, t);
}

// A-matrix (16x32 bf16) per-lane K offset of VGPR pair v (half = lane/16)
__device__ __forceinline__ int kA(int v, int hf) {
  return 2 * v + 16 * hf;
}

// wave32 xor-reductions via ds_swizzle (group-of-32 mode: and=0x1f, xor=m)
#define SWZF(x, m) __builtin_bit_cast(float, \
    __builtin_amdgcn_ds_swizzle(__builtin_bit_cast(int, (x)), 0x1f | ((m) << 10)))

__device__ __forceinline__ float wave_max(float x) {
  x = fmaxf(x, SWZF(x, 1));  x = fmaxf(x, SWZF(x, 2));
  x = fmaxf(x, SWZF(x, 4));  x = fmaxf(x, SWZF(x, 8));
  x = fmaxf(x, SWZF(x, 16));
  return x;
}
__device__ __forceinline__ float wave_sum(float x) {
  x += SWZF(x, 1);  x += SWZF(x, 2);  x += SWZF(x, 4);
  x += SWZF(x, 8);  x += SWZF(x, 16);
  return x;
}

// ---------------------------------------------------------------- convert
__global__ void __launch_bounds__(256)
cvt_bf16_kernel(const float* __restrict__ s, unsigned short* __restrict__ d, int n) {
  for (int i = blockIdx.x * blockDim.x + threadIdx.x; i < n;
       i += gridDim.x * blockDim.x)
    d[i] = f2bf(s[i]);
}

// ---------------------------------------------------------------- projection
// Y[b,h,s,d] = (X[n,:] @ W[o,:]^T + bias[o]) * outScale,  n=b*S+s, o=h*DK+d
// X: [B*S, DM] bf16   W: [DM, DM] bf16 (row-major, [out][in])
__global__ void __launch_bounds__(256)
proj_kernel(const unsigned short* __restrict__ X,
            const unsigned short* __restrict__ W,
            const float* __restrict__ bias,
            unsigned short* __restrict__ Y, float outScale) {
  const int wave = threadIdx.x >> 5;
  const int lane = threadIdx.x & 31;
  const int hf   = lane >> 4;
  const int r    = lane & 15;

  const int tile = blockIdx.x * 8 + wave;   // 4096 tiles total
  const int n0 = (tile >> 4) * 16;          // 256 row tiles
  const int o0 = (tile & 15) * 16;          // 16 col tiles

  v8f acc = {};
#pragma unroll
  for (int k0 = 0; k0 < DM; k0 += 32) {
    unsigned int au[8], bu[8];
#pragma unroll
    for (int v = 0; v < 8; ++v)
      au[v] = *(const unsigned int*)(X + (size_t)(n0 + r) * DM + k0 + kA(v, hf));
#pragma unroll
    for (int v = 0; v < 8; ++v)
      bu[v] = *(const unsigned int*)(W + (size_t)(o0 + r) * DM + k0 + 2 * v + 16 * hf);
    acc = __builtin_amdgcn_wmma_f32_16x16x32_bf16(
        false, mkfrag(au), false, mkfrag(bu), (short)0, acc, false, false);
  }

#pragma unroll
  for (int v = 0; v < 8; ++v) {
    const int n = n0 + v + 8 * hf;
    const int o = o0 + r;
    const float val = (acc[v] + bias[o]) * outScale;
    const int b = n >> 11, s = n & (SS - 1);
    const int h = o >> 5, d = o & (DK - 1);
    Y[((size_t)((b * HEADS + h) * SS + s)) * DK + d] = f2bf(val);
  }
}

// ---------------------------------------------------------------- attention
// One workgroup (8 waves) owns one (b,h, 16-query tile).
// LDS: scores[16][2048] fp32 (128KB) + O[16][32] fp32.
__global__ void __launch_bounds__(256)
attn_kernel(const unsigned short* __restrict__ Q,
            const unsigned short* __restrict__ K,
            const unsigned short* __restrict__ V,
            float* __restrict__ out) {
  extern __shared__ float smem[];
  float* sS = smem;                 // 16 * 2048
  float* sO = smem + 16 * SS;       // 16 * 32

  const int wave = threadIdx.x >> 5;
  const int lane = threadIdx.x & 31;
  const int hf   = lane >> 4;
  const int r    = lane & 15;

  const int qt = blockIdx.x & 127;       // 128 query tiles
  const int bh = blockIdx.x >> 7;        // 0..15
  const int q0 = qt * 16;
  const size_t base = (size_t)bh * SS;

  sO[threadIdx.x]       = 0.f;
  sO[threadIdx.x + 256] = 0.f;

  // -------- phase 1: scores S = (q * SCALE) . k^T  via WMMA --------
  unsigned int qu[8];
#pragma unroll
  for (int v = 0; v < 8; ++v)
    qu[v] = *(const unsigned int*)(Q + (base + q0 + r) * DK + kA(v, hf));
  const v16bf Aq = mkfrag(qu);

  for (int kt = wave; kt < SS / 16; kt += 8) {
    unsigned int ku[8];
#pragma unroll
    for (int v = 0; v < 8; ++v)
      ku[v] = *(const unsigned int*)(K + (base + kt * 16 + r) * DK + 2 * v + 16 * hf);
    v8f c = {};
    c = __builtin_amdgcn_wmma_f32_16x16x32_bf16(
        false, Aq, false, mkfrag(ku), (short)0, c, false, false);
#pragma unroll
    for (int v = 0; v < 8; ++v)
      sS[(v + 8 * hf) * SS + kt * 16 + r] = c[v];
    if (kt + 8 < SS / 16)
      __builtin_prefetch(K + (base + (kt + 8) * 16 + r) * DK, 0, 1);
  }
  __syncthreads();

  // -------- phase 2: sparsemax over the key dim (2048) per query row ------
  // tau solves sum(max(z - tau, 0)) == 1; bisection on [max-1, max].
  for (int t = 0; t < 2; ++t) {
    const int row = wave * 2 + t;
    float z[64];
#pragma unroll
    for (int e = 0; e < 64; ++e) z[e] = sS[row * SS + e * 32 + lane];
    float m = -3.4e38f;
#pragma unroll
    for (int e = 0; e < 64; ++e) m = fmaxf(m, z[e]);
    m = wave_max(m);
    float lo = m - 1.0f, hi = m;
    for (int it = 0; it < 40; ++it) {
      const float mid = 0.5f * (lo + hi);
      float s = 0.f;
#pragma unroll
      for (int e = 0; e < 64; ++e) s += fmaxf(z[e] - mid, 0.f);
      s = wave_sum(s);
      if (s > 1.0f) lo = mid; else hi = mid;
    }
    const float tau = 0.5f * (lo + hi);
#pragma unroll
    for (int e = 0; e < 64; ++e)
      sS[row * SS + e * 32 + lane] = fmaxf(z[e] - tau, 0.f);
  }
  __syncthreads();

  // -------- phase 3: O = P @ V via WMMA, keys split across 8 waves --------
  v8f acc0 = {}, acc1 = {};
#pragma unroll
  for (int st = 0; st < 8; ++st) {
    const int c0 = wave * 256 + st * 32;
    unsigned int au[8];
#pragma unroll
    for (int v = 0; v < 8; ++v) {
      const int kk = c0 + kA(v, hf);
      au[v] = (unsigned int)f2bf(sS[r * SS + kk]) |
              ((unsigned int)f2bf(sS[r * SS + kk + 1]) << 16);
    }
    const v16bf Ap = mkfrag(au);
    unsigned int bu0[8], bu1[8];
#pragma unroll
    for (int v = 0; v < 8; ++v) {
      const int key = c0 + 2 * v + 16 * hf;
      const unsigned short* vp0 = V + (base + key) * DK + r;
      const unsigned short* vp1 = V + (base + key + 1) * DK + r;
      bu0[v] = (unsigned int)vp0[0]  | ((unsigned int)vp1[0]  << 16);
      bu1[v] = (unsigned int)vp0[16] | ((unsigned int)vp1[16] << 16);
    }
    acc0 = __builtin_amdgcn_wmma_f32_16x16x32_bf16(
        false, Ap, false, mkfrag(bu0), (short)0, acc0, false, false);
    acc1 = __builtin_amdgcn_wmma_f32_16x16x32_bf16(
        false, Ap, false, mkfrag(bu1), (short)0, acc1, false, false);
  }
#pragma unroll
  for (int v = 0; v < 8; ++v) {
    const int m = v + 8 * hf;
    atomicAdd(&sO[m * DK + r],      acc0[v]);
    atomicAdd(&sO[m * DK + 16 + r], acc1[v]);
  }
  __syncthreads();

  // -------- write out[b, q0+m, h*DK + d] --------
  const int b = bh >> 3, h = bh & 7;
  for (int i = threadIdx.x; i < 16 * DK; i += 256) {
    const int m = i >> 5, d = i & (DK - 1);
    out[((size_t)(b * SS + q0 + m)) * DM + h * DK + d] = sO[i];
  }
}

// ---------------------------------------------------------------- launcher
extern "C" void kernel_launch(void* const* d_in, const int* in_sizes, int n_in,
                              void* d_out, int out_size, void* d_ws, size_t ws_size,
                              hipStream_t stream) {
  (void)in_sizes; (void)n_in; (void)out_size; (void)ws_size;
  const float* xq = (const float*)d_in[0];
  const float* xk = (const float*)d_in[1];
  const float* xv = (const float*)d_in[2];
  const float* Wq = (const float*)d_in[3];
  const float* bq = (const float*)d_in[4];
  const float* Wk = (const float*)d_in[5];
  const float* bk = (const float*)d_in[6];
  const float* Wv = (const float*)d_in[7];
  const float* bv = (const float*)d_in[8];

  const size_t NX = (size_t)BB * SS * DM;  // 1,048,576
  const size_t NW = (size_t)DM * DM;       // 65,536

  unsigned short* xqb = (unsigned short*)d_ws;
  unsigned short* xkb = xqb + NX;
  unsigned short* xvb = xkb + NX;
  unsigned short* wqb = xvb + NX;
  unsigned short* wkb = wqb + NW;
  unsigned short* wvb = wkb + NW;
  unsigned short* qb  = wvb + NW;   // [B,H,S,DK] each == NX elements
  unsigned short* kb  = qb + NX;
  unsigned short* vb  = kb + NX;

  cvt_bf16_kernel<<<512, 256, 0, stream>>>(xq, xqb, (int)NX);
  cvt_bf16_kernel<<<512, 256, 0, stream>>>(xk, xkb, (int)NX);
  cvt_bf16_kernel<<<512, 256, 0, stream>>>(xv, xvb, (int)NX);
  cvt_bf16_kernel<<<128, 256, 0, stream>>>(Wq, wqb, (int)NW);
  cvt_bf16_kernel<<<128, 256, 0, stream>>>(Wk, wkb, (int)NW);
  cvt_bf16_kernel<<<128, 256, 0, stream>>>(Wv, wvb, (int)NW);

  proj_kernel<<<512, 256, 0, stream>>>(xqb, wqb, bq, qb, QSCALE);
  proj_kernel<<<512, 256, 0, stream>>>(xkb, wkb, bk, kb, 1.0f);
  proj_kernel<<<512, 256, 0, stream>>>(xvb, wvb, bv, vb, 1.0f);

  const size_t shmem = (size_t)(16 * SS + 16 * DK) * sizeof(float); // 133,120 B
  attn_kernel<<<BB * HEADS * (SS / 16), 256, shmem, stream>>>(
      qb, kb, vb, (float*)d_out);
}
